// Decoder_3453153706557
// MI455X (gfx1250) — matrix operand order repaired and checked by reading, not compiled, not run on hardware
//
#include <hip/hip_runtime.h>
#include <hip/hip_bf16.h>
#include <math.h>

typedef __attribute__((ext_vector_type(16))) __bf16 v16bf;
typedef __attribute__((ext_vector_type(8)))  float  v8f;

#define DMODEL 1024
#define NHEADS 16
#define HDIM   64
#define FFDIM  4096
#define SEQ    2048
#define NBATCH 2
#define MROWS  (NBATCH * SEQ)

// ---------------------------------------------------------------------------
// helpers
// ---------------------------------------------------------------------------
static __device__ __forceinline__ unsigned short f32_to_bf16(float f) {
  unsigned u = __float_as_uint(f);
  u += 0x7FFFu + ((u >> 16) & 1u);           // round-to-nearest-even
  return (unsigned short)(u >> 16);
}

static __device__ __forceinline__ unsigned pack_bf16x2(float lo, float hi) {
  return ((unsigned)f32_to_bf16(hi) << 16) | (unsigned)f32_to_bf16(lo);
}

static __device__ __forceinline__ v16bf ld_frag(const unsigned short* p) {
  v16bf r;
  __builtin_memcpy(&r, p, sizeof(r));        // -> 2x ds_load_b128
  return r;
}

static __device__ __forceinline__ v8f wmma_bf16(v16bf a, v16bf b, v8f c) {
  // D(16x16,f32) = A(16x32,bf16) * B(32x16,bf16) + C
  return __builtin_amdgcn_wmma_f32_16x16x32_bf16(false, a, false, b,
                                                 (short)0, c, false, false);
}

static __device__ __forceinline__ float gelu_erf(float x) {
  return 0.5f * x * (1.0f + erff(x * 0.70710678118654752f));
}

// A-frag (16x32 bf16, MxK): element (r,k) -> lane = r + 16*k[4], slot = k & 15
// B-frag (32x16 bf16, KxN): element (k,c) -> lane = c + 16*k[4], slot = k & 15
// C-frag (16x16 f32):       element (m,n) -> lane = n + 16*(m>=8), vgpr = m & 7

// ---------------------------------------------------------------------------
// GEMM: C[M,N] = act(A[M,K] @ W[K,N] + bias[N]) ; ACT: 0=none, 1=gelu(erf)
// block tile 128x128x32, 256 threads = 8 waves in 4(M) x 2(N) layout.
// Software pipeline: global float4 prefetch of tile k+1 is issued before the
// 8 WMMAs of tile k, so HBM/L2 latency overlaps matrix math.
// ---------------------------------------------------------------------------
template<int ACT>
__global__ __launch_bounds__(256) void gemm_bias_kernel(
    const float* __restrict__ A, const float* __restrict__ W,
    const float* __restrict__ bias, float* __restrict__ C,
    int M, int N, int K)
{
  __shared__ alignas(32) unsigned short smA[8 * 32 * 16];  // 8 A-frags (16 rows each)
  __shared__ alignas(32) unsigned short smB[8 * 32 * 16];  // 8 B-frags (16 cols each)

  const int tid  = threadIdx.x;
  const int lane = tid & 31;
  const int wid  = tid >> 5;
  const int wM   = wid >> 1;              // 0..3
  const int wN   = wid & 1;               // 0..1
  const int m0   = blockIdx.y * 128;
  const int n0   = blockIdx.x * 128;

  // per-thread staging coordinates (16 elements of A + 16 of W per tile)
  int aRow[4], aKK[4], wKR[4], wCC[4];
  #pragma unroll
  for (int i = 0; i < 4; i++) {
    int s   = tid + i * 256;          // 0..1023 float4 slots
    aRow[i] = s >> 3;                 // 0..127
    aKK[i]  = (s & 7) << 2;           // 0..28
    wKR[i]  = s >> 5;                 // 0..31
    wCC[i]  = (s & 31) << 2;          // 0..124
  }

  const v8f zero = {0.f, 0.f, 0.f, 0.f, 0.f, 0.f, 0.f, 0.f};
  v8f acc[2][4];
  #pragma unroll
  for (int mi = 0; mi < 2; mi++)
    #pragma unroll
    for (int ni = 0; ni < 4; ni++) acc[mi][ni] = zero;

  // ---- prologue: prefetch tile 0 into registers ----------------------------
  float4 pa[4], pw[4];
  #pragma unroll
  for (int i = 0; i < 4; i++) {
    pa[i] = *(const float4*)(A + (size_t)(m0 + aRow[i]) * K + aKK[i]);
    pw[i] = *(const float4*)(W + (size_t)wKR[i] * N + n0 + wCC[i]);
  }

  for (int k0 = 0; k0 < K; k0 += 32) {
    __syncthreads();                       // prior iter finished reading LDS
    // -------- convert + store the prefetched tile into frag-order LDS -------
    #pragma unroll
    for (int i = 0; i < 4; i++) {
      {
        int frag = aRow[i] >> 4, r = aRow[i] & 15;
        float v4[4] = {pa[i].x, pa[i].y, pa[i].z, pa[i].w};
        #pragma unroll
        for (int j = 0; j < 4; j += 2) {   // pairs share lane, adjacent slots
          int k2 = aKK[i] + j;
          int ln = r + ((k2 >> 4) << 4);
          int sl = k2 & 15;
          *(unsigned*)&smA[(frag * 32 + ln) * 16 + sl] =
              pack_bf16x2(v4[j], v4[j + 1]);
        }
      }
      {
        int kr = wKR[i];
        float v4[4] = {pw[i].x, pw[i].y, pw[i].z, pw[i].w};
        #pragma unroll
        for (int j = 0; j < 4; j++) {      // adjacent cols -> adjacent lanes
          int c  = wCC[i] + j;
          int fr = c >> 4;
          int ln = (c & 15) + ((kr >> 4) << 4);
          smB[(fr * 32 + ln) * 16 + (kr & 15)] = f32_to_bf16(v4[j]);
        }
      }
    }
    __syncthreads();                       // tile k0 visible to all waves

    // -------- prefetch tile k0+32 (overlaps with the WMMAs below) -----------
    if (k0 + 32 < K) {
      #pragma unroll
      for (int i = 0; i < 4; i++) {
        pa[i] = *(const float4*)(A + (size_t)(m0 + aRow[i]) * K + (k0 + 32) + aKK[i]);
        pw[i] = *(const float4*)(W + (size_t)(k0 + 32 + wKR[i]) * N + n0 + wCC[i]);
      }
    }

    // -------- compute: 8 WMMAs per wave per k-tile --------------------------
    v16bf aF[2], bF[4];
    #pragma unroll
    for (int mi = 0; mi < 2; mi++)
      aF[mi] = ld_frag(&smA[((wM * 2 + mi) * 32 + lane) * 16]);
    #pragma unroll
    for (int ni = 0; ni < 4; ni++)
      bF[ni] = ld_frag(&smB[((wN * 4 + ni) * 32 + lane) * 16]);
    #pragma unroll
    for (int mi = 0; mi < 2; mi++)
      #pragma unroll
      for (int ni = 0; ni < 4; ni++)
        acc[mi][ni] = wmma_bf16(aF[mi], bF[ni], acc[mi][ni]);
  }

  // -------- epilogue ---------------------------------------------------------
  const int nl = lane & 15, hi = lane >> 4;
  #pragma unroll
  for (int mi = 0; mi < 2; mi++) {
    #pragma unroll
    for (int ni = 0; ni < 4; ni++) {
      int col = n0 + wN * 64 + ni * 16 + nl;
      float bv = bias[col];
      #pragma unroll
      for (int v = 0; v < 8; v++) {
        int row = m0 + wM * 32 + mi * 16 + v + (hi << 3);
        float val = acc[mi][ni][v] + bv;
        if (ACT == 1) val = gelu_erf(val);
        C[(size_t)row * N + col] = val;
      }
    }
  }
}

// ---------------------------------------------------------------------------
// Flash attention, causal. Q/K/V are [B*S, D] with head columns h*64..h*64+63.
// grid = (S/128, B*H); 256 threads = 8 waves, 16 query rows per wave.
// ---------------------------------------------------------------------------
__global__ __launch_bounds__(256) void attn_kernel(
    const float* __restrict__ Q, const float* __restrict__ K,
    const float* __restrict__ V, float* __restrict__ Op)
{
  __shared__ alignas(32) unsigned short smK[4 * 32 * 16]; // K^T frags [ks*2+nf]
  __shared__ alignas(32) unsigned short smV[4 * 32 * 16]; // V frags   [nf]
  __shared__ alignas(32) unsigned short smP[8 * 32 * 16]; // P A-frags per wave

  const int tid  = threadIdx.x;
  const int lane = tid & 31;
  const int wid  = tid >> 5;
  const int n    = lane & 15;
  const int hi   = lane >> 4;
  const int qb   = blockIdx.x;
  const int b    = blockIdx.y >> 4;
  const int h    = blockIdx.y & 15;

  const size_t rowbase = (size_t)b * SEQ;
  const int coff = h * HDIM;
  const int qs0  = qb * 128 + wid * 16;

  // ---- load Q A-fragments (pre-scaled by 1/sqrt(hd)) -----------------------
  const float* Qp = Q + (rowbase + qs0 + n) * DMODEL + coff;
  v16bf qf[2];
  #pragma unroll
  for (int ks = 0; ks < 2; ks++) {
    unsigned short t[16];
    #pragma unroll
    for (int sl = 0; sl < 16; sl++) {
      int hd = sl + (hi << 4) + (ks << 5);
      t[sl] = f32_to_bf16(Qp[hd] * 0.125f);
    }
    __builtin_memcpy(&qf[ks], t, 32);
  }

  const v8f zero = {0.f, 0.f, 0.f, 0.f, 0.f, 0.f, 0.f, 0.f};
  v8f o[4];
  #pragma unroll
  for (int nf = 0; nf < 4; nf++) o[nf] = zero;
  float mrun[8], lrun[8];
  #pragma unroll
  for (int v = 0; v < 8; v++) { mrun[v] = -1e30f; lrun[v] = 0.f; }

  const float* Kp = K + rowbase * DMODEL + coff;
  const float* Vp = V + rowbase * DMODEL + coff;

  const int nkb = (qb + 1) * 4;                // causal bound (32 keys/block)
  for (int kb = 0; kb < nkb; kb++) {
    const int key0 = kb * 32;
    __syncthreads();
    // ---- stage K^T and V blocks (32 keys x 64 hd) into frag-order LDS ------
    #pragma unroll
    for (int i = 0; i < 4; i++) {
      int idx = tid + i * 256;                 // 0..1023 float2 slots
      int kl  = idx >> 5;                      // key 0..31
      int hg  = (idx & 31) << 1;               // hd 0,2,..62
      float2 kv = *(const float2*)(Kp + (size_t)(key0 + kl) * DMODEL + hg);
      int fk = ((hg >> 5) << 1) + (kl >> 4);
      int lk = (kl & 15) + (((hg >> 4) & 1) << 4);
      *(unsigned*)&smK[(fk * 32 + lk) * 16 + (hg & 15)] = pack_bf16x2(kv.x, kv.y);
      float2 vv = *(const float2*)(Vp + (size_t)(key0 + kl) * DMODEL + hg);
      int fv = hg >> 4;                        // same for hg, hg+1
      int lv = (hg & 15) + ((kl >> 4) << 4);   // adjacent hd -> adjacent lanes
      smV[(fv * 32 + lv) * 16 + (kl & 15)]       = f32_to_bf16(vv.x);
      smV[(fv * 32 + lv + 1) * 16 + (kl & 15)]   = f32_to_bf16(vv.y);
    }
    __syncthreads();

    // ---- scores: S(16x32) = Q(16x64) @ K^T(64x32) --------------------------
    v16bf kf00 = ld_frag(&smK[(0 * 32 + lane) * 16]);
    v16bf kf01 = ld_frag(&smK[(1 * 32 + lane) * 16]);
    v16bf kf10 = ld_frag(&smK[(2 * 32 + lane) * 16]);
    v16bf kf11 = ld_frag(&smK[(3 * 32 + lane) * 16]);
    v8f s0 = zero, s1 = zero;
    s0 = wmma_bf16(qf[0], kf00, s0);
    s0 = wmma_bf16(qf[1], kf10, s0);
    s1 = wmma_bf16(qf[0], kf01, s1);
    s1 = wmma_bf16(qf[1], kf11, s1);

    v16bf vf[4];
    #pragma unroll
    for (int nf = 0; nf < 4; nf++) vf[nf] = ld_frag(&smV[(nf * 32 + lane) * 16]);

    // ---- online softmax (rows live across the 16-lane half-wave) -----------
    #pragma unroll
    for (int v = 0; v < 8; v++) {
      float a0 = s0[v], a1 = s1[v];
      int qrow = qs0 + v + (hi << 3);
      if (key0 + n      > qrow) a0 = -1e30f;
      if (key0 + 16 + n > qrow) a1 = -1e30f;
      float mx = fmaxf(a0, a1);
      mx = fmaxf(mx, __shfl_xor(mx, 1, 32));
      mx = fmaxf(mx, __shfl_xor(mx, 2, 32));
      mx = fmaxf(mx, __shfl_xor(mx, 4, 32));
      mx = fmaxf(mx, __shfl_xor(mx, 8, 32));
      float mnew  = fmaxf(mrun[v], mx);
      float alpha = __expf(mrun[v] - mnew);
      float p0 = __expf(a0 - mnew);
      float p1 = __expf(a1 - mnew);
      float sum = p0 + p1;
      sum += __shfl_xor(sum, 1, 32);
      sum += __shfl_xor(sum, 2, 32);
      sum += __shfl_xor(sum, 4, 32);
      sum += __shfl_xor(sum, 8, 32);
      lrun[v] = lrun[v] * alpha + sum;
      mrun[v] = mnew;
      #pragma unroll
      for (int nf = 0; nf < 4; nf++) o[nf][v] *= alpha;
      // scatter P (C-layout) into A-frag layout in per-wave LDS scratch
      int m = v + (hi << 3);
      int k0c = n;
      smP[(wid * 32 + m + ((k0c >> 4) << 4)) * 16 +
          (k0c & 15)] = f32_to_bf16(p0);
      int k1c = n + 16;
      smP[(wid * 32 + m + ((k1c >> 4) << 4)) * 16 +
          (k1c & 15)] = f32_to_bf16(p1);
    }
    __syncthreads();

    // ---- O(16x64) += P(16x32) @ V(32x64) -----------------------------------
    v16bf pf = ld_frag(&smP[(wid * 32 + lane) * 16]);
    #pragma unroll
    for (int nf = 0; nf < 4; nf++) o[nf] = wmma_bf16(pf, vf[nf], o[nf]);
  }

  // ---- normalize and write ctx in [b, s, h*64+hd] layout --------------------
  #pragma unroll
  for (int nf = 0; nf < 4; nf++) {
    #pragma unroll
    for (int v = 0; v < 8; v++) {
      int qrow  = qs0 + v + (hi << 3);
      float val = o[nf][v] * (1.0f / lrun[v]);
      Op[(rowbase + qrow) * DMODEL + coff + nf * 16 + n] = val;
    }
  }
}

// ---------------------------------------------------------------------------
// out[row] = LayerNorm(a[row] + res[row]) * g + b     (D = 1024, 256 threads)
// ---------------------------------------------------------------------------
__global__ __launch_bounds__(256) void ln_residual_kernel(
    const float* __restrict__ a, const float* __restrict__ res,
    const float* __restrict__ g, const float* __restrict__ bta,
    float* __restrict__ out)
{
  const int row  = blockIdx.x;
  const int tid  = threadIdx.x;
  const int lane = tid & 31;
  const int wid  = tid >> 5;
  __shared__ float red1[8];
  __shared__ float red2[8];

  const float4* a4 = (const float4*)(a   + (size_t)row * DMODEL);
  const float4* r4 = (const float4*)(res + (size_t)row * DMODEL);
  float4 x = a4[tid];
  float4 r = r4[tid];
  x.x += r.x; x.y += r.y; x.z += r.z; x.w += r.w;

  float s = x.x + x.y + x.z + x.w;
  s += __shfl_xor(s, 1, 32);  s += __shfl_xor(s, 2, 32);
  s += __shfl_xor(s, 4, 32);  s += __shfl_xor(s, 8, 32);
  s += __shfl_xor(s, 16, 32);
  if (lane == 0) red1[wid] = s;
  __syncthreads();
  float tot = red1[0] + red1[1] + red1[2] + red1[3] +
              red1[4] + red1[5] + red1[6] + red1[7];
  float mu = tot * (1.0f / DMODEL);

  float d0 = x.x - mu, d1 = x.y - mu, d2 = x.z - mu, d3 = x.w - mu;
  float sq = d0 * d0 + d1 * d1 + d2 * d2 + d3 * d3;
  sq += __shfl_xor(sq, 1, 32);  sq += __shfl_xor(sq, 2, 32);
  sq += __shfl_xor(sq, 4, 32);  sq += __shfl_xor(sq, 8, 32);
  sq += __shfl_xor(sq, 16, 32);
  if (lane == 0) red2[wid] = sq;
  __syncthreads();
  float var = (red2[0] + red2[1] + red2[2] + red2[3] +
               red2[4] + red2[5] + red2[6] + red2[7]) * (1.0f / DMODEL);
  float inv = rsqrtf(var + 1e-5f);

  const float4 g4 = ((const float4*)g)[tid];
  const float4 b4 = ((const float4*)bta)[tid];
  float4 o;
  o.x = d0 * inv * g4.x + b4.x;
  o.y = d1 * inv * g4.y + b4.y;
  o.z = d2 * inv * g4.z + b4.z;
  o.w = d3 * inv * g4.w + b4.w;
  ((float4*)(out + (size_t)row * DMODEL))[tid] = o;
}

// ---------------------------------------------------------------------------
// host launch
// ---------------------------------------------------------------------------
extern "C" void kernel_launch(void* const* d_in, const int* in_sizes, int n_in,
                              void* d_out, int out_size, void* d_ws, size_t ws_size,
                              hipStream_t stream) {
  (void)in_sizes; (void)n_in; (void)out_size; (void)ws_size;
  const float* x   = (const float*)d_in[0];
  const float* Wq  = (const float*)d_in[1];
  const float* bq  = (const float*)d_in[2];
  const float* Wk  = (const float*)d_in[3];
  const float* bk  = (const float*)d_in[4];
  const float* Wv  = (const float*)d_in[5];
  const float* bv  = (const float*)d_in[6];
  const float* Wo  = (const float*)d_in[7];
  const float* bo  = (const float*)d_in[8];
  const float* g1  = (const float*)d_in[9];
  const float* be1 = (const float*)d_in[10];
  const float* W1  = (const float*)d_in[11];
  const float* bb1 = (const float*)d_in[12];
  const float* W2  = (const float*)d_in[13];
  const float* bb2 = (const float*)d_in[14];
  const float* g2  = (const float*)d_in[15];
  const float* be2 = (const float*)d_in[16];
  float* out = (float*)d_out;

  const size_t SZ = (size_t)MROWS * DMODEL;     // 4M floats (16 MB)
  float* ws  = (float*)d_ws;
  float* q   = ws;
  float* k   = ws + SZ;
  float* v   = ws + 2 * SZ;
  float* ctx = ws + 3 * SZ;
  float* tmp = ws + 4 * SZ;
  float* hh  = ws + 5 * SZ;
  float* h1  = ws;                               // 64 MB, reuses q..ctx (dead)

  dim3 blk(256);
  dim3 gProj(DMODEL / 128, MROWS / 128);         // (8, 32)
  gemm_bias_kernel<0><<<gProj, blk, 0, stream>>>(x, Wq, bq, q, MROWS, DMODEL, DMODEL);
  gemm_bias_kernel<0><<<gProj, blk, 0, stream>>>(x, Wk, bk, k, MROWS, DMODEL, DMODEL);
  gemm_bias_kernel<0><<<gProj, blk, 0, stream>>>(x, Wv, bv, v, MROWS, DMODEL, DMODEL);

  dim3 gAttn(SEQ / 128, NBATCH * NHEADS);        // (16, 32)
  attn_kernel<<<gAttn, blk, 0, stream>>>(q, k, v, ctx);

  gemm_bias_kernel<0><<<gProj, blk, 0, stream>>>(ctx, Wo, bo, tmp, MROWS, DMODEL, DMODEL);
  ln_residual_kernel<<<dim3(MROWS), blk, 0, stream>>>(tmp, x, g1, be1, hh);

  dim3 gFF1(FFDIM / 128, MROWS / 128);           // (32, 32)
  gemm_bias_kernel<1><<<gFF1, blk, 0, stream>>>(hh, W1, bb1, h1, MROWS, FFDIM, DMODEL);
  gemm_bias_kernel<0><<<gProj, blk, 0, stream>>>(h1, W2, bb2, tmp, MROWS, DMODEL, FFDIM);
  ln_residual_kernel<<<dim3(MROWS), blk, 0, stream>>>(tmp, hh, g2, be2, out);
}
